// StateSpaceModel_29300266893417
// MI455X (gfx1250) — matrix operand order repaired and checked by reading, not compiled, not run on hardware
//
#include <hip/hip_runtime.h>

typedef __attribute__((ext_vector_type(16))) __bf16 v16bf;
typedef __attribute__((ext_vector_type(8)))  float  v8f;

#define BATCH   2
#define SEQ     2048
#define DIN     2048
#define DSTATE  16
#define DTRANK  128
#define EPROJ   160                 // DTRANK + 2*DSTATE
#define MTOT    (BATCH * SEQ)       // 4096
#define CH      128                 // scan chunk length
#define NCH     (SEQ / CH)          // 16 chunks

// ---------------------------------------------------------------------------
// GEMM1: xdbl[m, e] = sum_k x[m, k] * xw[e, k]   (M=4096, N=160, K=2048)
// 8 waves/block = 4 M-tiles x 2 E-halves. Each wave holds FIVE 16x16 f32
// accumulators (E-tiles eg*5 .. eg*5+4), so one A fragment feeds 5 WMMAs:
// x is read once per E-half instead of once per E-tile.
// A lane layout (16x32 bf16): lane l16, K = k0 + half*16 + {0..15}
// B lane layout (32x16 bf16): col l16, K = k0 + half*16 + {0..15}
// C/D f32 layout: VGPR r -> row m0 + half*8 + r, col e0 + l16
// ---------------------------------------------------------------------------
__global__ __launch_bounds__(256) void gemm1_kernel(const float* __restrict__ x,
                                                    const float* __restrict__ xw,
                                                    float* __restrict__ xdbl) {
  const int wave = threadIdx.x >> 5;           // 0..7
  const int lane = threadIdx.x & 31;
  const int mt   = blockIdx.x * 4 + (wave >> 1);
  const int eg   = wave & 1;                   // E half
  const int m0   = mt * 16;
  const int half = lane >> 4;
  const int l16  = lane & 15;

  const float* arow = x + (size_t)(m0 + l16) * DIN;

  v8f acc[5] = {};
  for (int k0 = 0; k0 < DIN; k0 += 32) {
    const float* ap = arow + k0 + half * 16;
    __builtin_prefetch(ap + 64, 0, 0);         // global_prefetch_b8: K tile two ahead
    v16bf a;
#pragma unroll
    for (int i = 0; i < 8; ++i) {
      a[i]     = (__bf16)ap[i];
      a[i + 8] = (__bf16)ap[8 + i];
    }
#pragma unroll
    for (int j = 0; j < 5; ++j) {
      const float* bp = xw + (size_t)(eg * 80 + j * 16 + l16) * DIN + k0 + half * 16;
      v16bf b;
#pragma unroll
      for (int i = 0; i < 16; ++i) b[i] = (__bf16)bp[i];
      acc[j] = __builtin_amdgcn_wmma_f32_16x16x32_bf16(false, a, false, b,
                                                       (short)0, acc[j], false, false);
    }
  }
#pragma unroll
  for (int j = 0; j < 5; ++j)
#pragma unroll
    for (int r = 0; r < 8; ++r)
      xdbl[(size_t)(m0 + half * 8 + r) * EPROJ + (eg * 80 + j * 16 + l16)] = acc[j][r];
}

// ---------------------------------------------------------------------------
// GEMM2: delta[m, d] = softplus( sum_r xdbl[m, r] * dtw[d, r] + dtb[d] )
// M=4096, N=2048, K=128. One wave per 16x16 tile, K fully unrolled (4 WMMA).
// ---------------------------------------------------------------------------
__global__ __launch_bounds__(256) void gemm2_kernel(const float* __restrict__ xdbl,
                                                    const float* __restrict__ dtw,
                                                    const float* __restrict__ dtb,
                                                    float* __restrict__ delta) {
  const int wave = threadIdx.x >> 5;
  const int lane = threadIdx.x & 31;
  const int tile = blockIdx.x * 8 + wave;
  const int NTN  = DIN / 16;            // 128
  const int m0   = (tile / NTN) * 16;
  const int d0   = (tile % NTN) * 16;
  const int half = lane >> 4;
  const int l16  = lane & 15;

  const float* arow = xdbl + (size_t)(m0 + l16) * EPROJ;   // first DTRANK cols
  const float* brow = dtw  + (size_t)(d0 + l16) * DTRANK;

  v8f acc = {};
#pragma unroll
  for (int k0 = 0; k0 < DTRANK; k0 += 32) {
    const float* ap = arow + k0 + half * 16;
    const float* bp = brow + k0 + half * 16;
    v16bf a, b;
#pragma unroll
    for (int i = 0; i < 8; ++i) {
      a[i]     = (__bf16)ap[i];
      a[i + 8] = (__bf16)ap[8 + i];
    }
#pragma unroll
    for (int i = 0; i < 16; ++i) b[i] = (__bf16)bp[i];
    acc = __builtin_amdgcn_wmma_f32_16x16x32_bf16(false, a, false, b,
                                                  (short)0, acc, false, false);
  }
#pragma unroll
  for (int r = 0; r < 8; ++r) {
    const int m = m0 + half * 8 + r;
    const int d = d0 + l16;
    const float v  = acc[r] + dtb[d];
    const float sp = (v > 20.f) ? v : log1pf(__expf(v));   // softplus
    delta[(size_t)m * DIN + d] = sp;
  }
}

// ---------------------------------------------------------------------------
// Async-stage the chunk's B/C slab (CH rows x 32 floats = 16 KB) into LDS.
// All 256 threads of a block share (b, chunk), so the slab is block-uniform.
// Uses global_load_async_to_lds_b128 (ASYNCcnt) + s_wait_asynccnt + barrier.
// LDS layout: sBC[t*32 + n] = xdbl[b, c*CH + t, DTRANK + n]
// ---------------------------------------------------------------------------
__device__ __forceinline__ void stage_bc_slab(float* sBC, const float* bc) {
  const unsigned sbase = (unsigned)(size_t)sBC;
#pragma unroll
  for (int it = 0; it < 4; ++it) {
    const int s   = threadIdx.x + it * 256;    // 1024 x 16B segments
    const int t   = s >> 3;
    const int seg = s & 7;
    const unsigned ldsoff = sbase + (unsigned)s * 16;          // t*128 + seg*16
    const unsigned long long ga =
        (unsigned long long)(size_t)(bc + (size_t)t * EPROJ + seg * 4);
    asm volatile("global_load_async_to_lds_b128 %0, %1, off"
                 :: "v"(ldsoff), "v"(ga) : "memory");
  }
  asm volatile("s_wait_asynccnt 0" ::: "memory");
  __syncthreads();
}

// ---------------------------------------------------------------------------
// Scan pass 1: per-(b,d,chunk) composed diagonal transition:
//   h_out = P * h_in + q,  P[n] = prod exp(dt*A[n]),  q = a*q + (dt*u)*B[n].
// cs layout: [b][c][0..15]=P, [16..31]=q, innermost stride DIN over d.
// ---------------------------------------------------------------------------
__global__ __launch_bounds__(256) void scan_chunk_kernel(const float* __restrict__ A_log,
                                                         const float* __restrict__ x,
                                                         const float* __restrict__ delta,
                                                         const float* __restrict__ xdbl,
                                                         float* __restrict__ cs) {
  __shared__ float sBC[CH * 32];                              // 16 KB
  const int tid  = blockIdx.x * blockDim.x + threadIdx.x;     // BATCH*NCH*DIN
  const int d    = tid & (DIN - 1);
  const int rest = tid >> 11;        // DIN = 2^11
  const int c    = rest & (NCH - 1);
  const int b    = rest >> 4;        // NCH = 16

  const size_t row = (size_t)b * SEQ + (size_t)c * CH;
  const float* bc = xdbl + row * EPROJ + DTRANK;
  stage_bc_slab(sBC, bc);

  float A[DSTATE], P[DSTATE], q[DSTATE];
#pragma unroll
  for (int n = 0; n < DSTATE; ++n) {
    A[n] = -__expf(A_log[d * DSTATE + n]);
    P[n] = 1.f;
    q[n] = 0.f;
  }

  const float* dl = delta + row * DIN + d;
  const float* xu = x     + row * DIN + d;

  for (int t = 0; t < CH; ++t) {
    const float dt = dl[(size_t)t * DIN];
    const float u  = xu[(size_t)t * DIN];
    const float du = dt * u;
    const float* Bt = &sBC[t * 32];
#pragma unroll
    for (int n = 0; n < DSTATE; ++n) {
      const float a = __expf(dt * A[n]);
      P[n] *= a;
      q[n] = a * q[n] + du * Bt[n];
    }
  }

  float* base = cs + (size_t)(b * NCH + c) * 32 * DIN + d;
#pragma unroll
  for (int n = 0; n < DSTATE; ++n) {
    base[(size_t)n * DIN]        = P[n];
    base[(size_t)(n + 16) * DIN] = q[n];
  }
}

// ---------------------------------------------------------------------------
// Scan pass 2: serial combine over the 16 chunks per (b,d); records the
// state at each chunk entry. Only BATCH*DIN = 4096 threads, 16 cheap steps.
// ---------------------------------------------------------------------------
__global__ __launch_bounds__(256) void scan_combine_kernel(const float* __restrict__ cs,
                                                           float* __restrict__ hinit) {
  const int tid = blockIdx.x * blockDim.x + threadIdx.x;    // BATCH*DIN
  const int d   = tid & (DIN - 1);
  const int b   = tid >> 11;

  float h[DSTATE] = {};
  for (int c = 0; c < NCH; ++c) {
    float*       hb  = hinit + (size_t)(b * NCH + c) * DSTATE * DIN + d;
    const float* csb = cs    + (size_t)(b * NCH + c) * 32 * DIN + d;
#pragma unroll
    for (int n = 0; n < DSTATE; ++n) {
      hb[(size_t)n * DIN] = h[n];
      h[n] = csb[(size_t)n * DIN] * h[n] + csb[(size_t)(n + 16) * DIN];
    }
  }
}

// ---------------------------------------------------------------------------
// Scan pass 3: re-run each chunk from its known entry state and emit y.
// ---------------------------------------------------------------------------
__global__ __launch_bounds__(256) void scan_emit_kernel(const float* __restrict__ A_log,
                                                        const float* __restrict__ Dvec,
                                                        const float* __restrict__ x,
                                                        const float* __restrict__ delta,
                                                        const float* __restrict__ xdbl,
                                                        const float* __restrict__ hinit,
                                                        float* __restrict__ out) {
  __shared__ float sBC[CH * 32];                              // 16 KB
  const int tid  = blockIdx.x * blockDim.x + threadIdx.x;
  const int d    = tid & (DIN - 1);
  const int rest = tid >> 11;
  const int c    = rest & (NCH - 1);
  const int b    = rest >> 4;

  const size_t row = (size_t)b * SEQ + (size_t)c * CH;
  const float* bc = xdbl + row * EPROJ + DTRANK;
  stage_bc_slab(sBC, bc);

  float A[DSTATE], h[DSTATE];
  const float* hb = hinit + (size_t)(b * NCH + c) * DSTATE * DIN + d;
#pragma unroll
  for (int n = 0; n < DSTATE; ++n) {
    A[n] = -__expf(A_log[d * DSTATE + n]);
    h[n] = hb[(size_t)n * DIN];
  }
  const float Dd = Dvec[d];

  const float* dl = delta + row * DIN + d;
  const float* xu = x     + row * DIN + d;
  float*       ob = out   + row * DIN + d;

  for (int t = 0; t < CH; ++t) {
    const float dt = dl[(size_t)t * DIN];
    const float u  = xu[(size_t)t * DIN];
    const float du = dt * u;
    const float* Bt = &sBC[t * 32];
    const float* Ct = Bt + DSTATE;
    float y = 0.f;
#pragma unroll
    for (int n = 0; n < DSTATE; ++n) {
      const float a = __expf(dt * A[n]);
      h[n] = a * h[n] + du * Bt[n];
      y += h[n] * Ct[n];
    }
    ob[(size_t)t * DIN] = y + u * Dd;
  }
}

// ---------------------------------------------------------------------------
extern "C" void kernel_launch(void* const* d_in, const int* in_sizes, int n_in,
                              void* d_out, int out_size, void* d_ws, size_t ws_size,
                              hipStream_t stream) {
  (void)in_sizes; (void)n_in; (void)out_size; (void)ws_size;
  const float* x     = (const float*)d_in[0];
  const float* A_log = (const float*)d_in[1];
  const float* Dvec  = (const float*)d_in[2];
  const float* xpw   = (const float*)d_in[3];
  const float* dtw   = (const float*)d_in[4];
  const float* dtb   = (const float*)d_in[5];
  float* out = (float*)d_out;

  // Workspace carve-up (floats): all regions fully written before read.
  float* ws    = (float*)d_ws;
  float* xdbl  = ws;                                        // 4096*160
  float* delta = xdbl  + (size_t)MTOT * EPROJ;              // 4096*2048
  float* cs    = delta + (size_t)MTOT * DIN;                // 2*16*32*2048
  float* hinit = cs    + (size_t)BATCH * NCH * 32 * DIN;    // 2*16*16*2048

  // GEMM1: 256 M-tiles, 4 per block (8 waves: 4M x 2E-halves) -> 64 blocks
  gemm1_kernel<<<MTOT / 16 / 4, 256, 0, stream>>>(x, xpw, xdbl);
  // GEMM2: 32768 wave-tiles, 8 waves/block -> 4096 blocks
  gemm2_kernel<<<(MTOT / 16) * (DIN / 16) / 8, 256, 0, stream>>>(xdbl, dtw, dtb, delta);
  // Chunked scan: 65536 threads, then 4096-thread combine, then 65536-thread emit
  scan_chunk_kernel<<<BATCH * NCH * DIN / 256, 256, 0, stream>>>(A_log, x, delta, xdbl, cs);
  scan_combine_kernel<<<BATCH * DIN / 256, 256, 0, stream>>>(cs, hinit);
  scan_emit_kernel<<<BATCH * NCH * DIN / 256, 256, 0, stream>>>(A_log, Dvec, x, delta, xdbl, hinit, out);
}
